// SpectralAttentionBlock_83124797046955
// MI455X (gfx1250) — matrix operand order repaired and checked by reading, not compiled, not run on hardware
//
#include <hip/hip_runtime.h>
#include <hip/hip_bf16.h>

// ---------------------------------------------------------------------------
// SpectralAttentionBlock on gfx1250 (MI455X): WMMA f32_16x16x32_f16 for all
// dense GEMMs + flash attention; VALU/atomics for the sparse Chebyshev prop.
// GEMM: A pre-converted to f16; B staged through LDS (one 4KB tile shared by
// all 8 waves, register-staged double buffer). Final k-step peeled so the
// inner loop has no guards/selects -- straight-line load/WMMA/barrier flow.
// ---------------------------------------------------------------------------

typedef __attribute__((ext_vector_type(16))) _Float16 v16h;
typedef __attribute__((ext_vector_type(8)))  _Float16 v8h;
typedef __attribute__((ext_vector_type(8)))  float    v8f;

static constexpr int NN = 4096;   // nodes
static constexpr int DD = 512;    // feature dim
static constexpr int EE = 131072; // edges
static constexpr int KC = 12;     // cheb order
static constexpr int HH = 16;     // heads (head dim = 32)

__device__ __forceinline__ v8f wmma_f16(v16h a, v16h b, v8f c) {
  return __builtin_amdgcn_wmma_f32_16x16x32_f16(false, a, false, b, (short)0, c,
                                                false, false);
}

// ---------------------------- small kernels --------------------------------

__global__ void reg_loss_kernel(const float* __restrict__ x,
                                const int* __restrict__ row,
                                const int* __restrict__ col,
                                float* __restrict__ out) {
  int e = blockIdx.x * blockDim.x + threadIdx.x;
  if (e >= EE) return;
  const float* a = x + (size_t)row[e] * DD;
  const float* b = x + (size_t)col[e] * DD;
  float s = 0.f;
  for (int i = 0; i < DD; ++i) { float d = a[i] - b[i]; s += d * d; }
  atomicAdd(out, 1e-4f * s);
}

__global__ void deg_kernel(const int* __restrict__ row, float* __restrict__ deg) {
  int e = blockIdx.x * blockDim.x + threadIdx.x;
  if (e >= EE) return;
  atomicAdd(deg + row[e], 1.0f);
}

__global__ void dinv_kernel(const float* __restrict__ deg, float* __restrict__ dinv) {
  int i = blockIdx.x * blockDim.x + threadIdx.x;
  if (i >= NN) return;
  float d = deg[i];
  dinv[i] = (d > 0.f) ? rsqrtf(d) : 0.f;
}

__global__ void edge_norm_kernel(const int* __restrict__ row,
                                 const int* __restrict__ col,
                                 const float* __restrict__ dinv,
                                 float* __restrict__ nrm) {
  int e = blockIdx.x * blockDim.x + threadIdx.x;
  if (e >= EE) return;
  nrm[e] = dinv[row[e]] * dinv[col[e]];
}

// O[col[e]] += scale * nrm[e] * T[row[e]]   (per edge, 4-float chunks)
__global__ void prop_kernel(const float* __restrict__ T, float* __restrict__ O,
                            const int* __restrict__ row, const int* __restrict__ col,
                            const float* __restrict__ nrm, float scale) {
  int tid = blockIdx.x * blockDim.x + threadIdx.x;
  int e = tid >> 7;                 // D/4 = 128 chunks per edge
  if (e >= EE) return;
  int c = (tid & 127) << 2;
  float w = scale * nrm[e];
  if (w == 0.f) return;
  float4 t = *(const float4*)(T + (size_t)row[e] * DD + c);
  float* op = O + (size_t)col[e] * DD + c;
  atomicAdd(op + 0, w * t.x);
  atomicAdd(op + 1, w * t.y);
  atomicAdd(op + 2, w * t.z);
  atomicAdd(op + 3, w * t.w);
}

__global__ void neg_copy_kernel(const float* __restrict__ src, float* __restrict__ dst,
                                int n) {
  int i = blockIdx.x * blockDim.x + threadIdx.x;
  if (i < n) dst[i] = -src[i];
}

__global__ void add_kernel(const float* __restrict__ a, const float* __restrict__ b,
                           float* __restrict__ c, int n) {
  int i = blockIdx.x * blockDim.x + threadIdx.x;
  if (i < n) c[i] = a[i] + b[i];
}

__global__ void cvt_f16_kernel(const float* __restrict__ src, _Float16* __restrict__ dst,
                               int n) {
  int i = blockIdx.x * blockDim.x + threadIdx.x;
  if (i < n) dst[i] = (_Float16)src[i];
}

// vt[(head*32+hd)*N + n] = qkv[n*1536 + 1024 + head*32 + hd]
__global__ void v_transpose_kernel(const float* __restrict__ qkv, _Float16* __restrict__ vt) {
  int i = blockIdx.x * blockDim.x + threadIdx.x;   // N*512 threads
  if (i >= NN * 512) return;
  int n = i >> 9;
  int j = i & 511;
  vt[(size_t)j * NN + n] = (_Float16)qkv[(size_t)n * 1536 + 1024 + j];
}

// row-wise layernorm, one block (128 thr) per row
__global__ void ln_kernel(const float* __restrict__ X, const float* __restrict__ g,
                          const float* __restrict__ b, float* __restrict__ Y) {
  int r = blockIdx.x, t = threadIdx.x;
  const float* xr = X + (size_t)r * DD;
  float s = 0.f, s2 = 0.f;
  for (int i = t; i < DD; i += 128) { float v = xr[i]; s += v; s2 += v * v; }
  __shared__ float rs[128], rq[128];
  rs[t] = s; rq[t] = s2;
  __syncthreads();
  for (int o = 64; o > 0; o >>= 1) {
    if (t < o) { rs[t] += rs[t + o]; rq[t] += rq[t + o]; }
    __syncthreads();
  }
  float mu = rs[0] * (1.f / DD);
  float var = rq[0] * (1.f / DD) - mu * mu;
  float inv = rsqrtf(var + 1e-5f);
  for (int i = t; i < DD; i += 128)
    Y[(size_t)r * DD + i] = (xr[i] - mu) * inv * g[i] + b[i];
}

// Pack fp32 weight into WMMA B-fragment order (f16).
// B[k][n] = transW ? W[n*ldw + k] : W[k*ldw + n]
// Bp[((kt*(Nn/16)+nt)*32 + lane)*16 + i], lane: n = nt*16+(lane&15),
//                                        k = kt*32 + (lane>>4)*8 + i + (i&8)
__global__ void pack_b_kernel(const float* __restrict__ W, _Float16* __restrict__ Bp,
                              int Kk, int Nn, int ldw, int transW) {
  int t = blockIdx.x * blockDim.x + threadIdx.x;
  int nnt = Nn >> 4;
  int total = (Kk >> 5) * nnt * 32;
  if (t >= total) return;
  int lane = t & 31;
  int tile = t >> 5;
  int nt = tile % nnt, kt = tile / nnt;
  int n = nt * 16 + (lane & 15);
  int kbase = kt * 32 + ((lane >> 4) << 3);
  _Float16* dst = Bp + ((size_t)tile * 32 + lane) * 16;
  for (int i = 0; i < 16; ++i) {
    int k = kbase + i + (i & 8);
    float w = transW ? W[(size_t)n * ldw + k] : W[(size_t)k * ldw + n];
    dst[i] = (_Float16)w;
  }
}

// ------------------------------ WMMA GEMM ----------------------------------
// C[M,Nn] (fp32) = A[M,Kk] (f16, pre-converted) @ Bp (pre-packed f16)
// 8 waves / block; wave computes 16 rows x 64 cols; grid (M/128, Nn/64).
// Per k-step all 8 waves share one 4KB B tile staged through LDS:
//   global_load (into regs, early) -> WMMAs on previous stage -> ds_store
//   (late) -> barrier.  Final k-step peeled: no guards in the hot loop.
#define GF_ACC  1
#define GF_BIAS 2
#define GF_GELU 4
#define GF_RES  8

__global__ void gemm_wmma_kernel(const _Float16* __restrict__ A,
                                 const _Float16* __restrict__ Bp,
                                 const float* __restrict__ bias,
                                 const float* __restrict__ Res,
                                 float* __restrict__ C,
                                 int M, int Nn, int Kk, int flags) {
  __shared__ v8h Bs[2][256];           // 2 stages x 4KB
  int tid = threadIdx.x;
  int lane = tid & 31;
  int w = tid >> 5;
  int m0 = blockIdx.x * 128 + w * 16;
  int n0 = blockIdx.y * 64;
  int hi8 = (lane >> 4) << 3;
  int l15 = lane & 15;
  int nnt = Nn >> 4;
  const _Float16* arow = A + (size_t)(m0 + l15) * Kk;

  // cooperative B stage source: 4 consecutive 16-col tiles = 2048 halves
  auto bsrc = [&](int k0) -> const v8h* {
    return (const v8h*)(Bp + ((size_t)(k0 >> 5) * nnt + (n0 >> 4)) * 512) + tid;
  };

  v8f acc0 = {}, acc1 = {}, acc2 = {}, acc3 = {};

  // prologue: stage 0 into LDS, A fragment for k0 = 0
  Bs[0][tid] = *bsrc(0);
  v8h a_lo = *(const v8h*)(arow + hi8);
  v8h a_hi = *(const v8h*)(arow + 16 + hi8);
  __syncthreads();

  int s = 0;
  int k0 = 0;
  for (; k0 + 32 < Kk; k0 += 32, s ^= 1) {
    // issue next-stage global load early (lands in regs, stored after WMMAs)
    v8h stage = *bsrc(k0 + 32);

    // assemble A fragment, then dead-write reload for next k-step
    v16h a;
    for (int i = 0; i < 8; ++i) { a[i] = a_lo[i]; a[8 + i] = a_hi[i]; }
    a_lo = *(const v8h*)(arow + k0 + 32 + hi8);
    a_hi = *(const v8h*)(arow + k0 + 48 + hi8);

    // B fragments from current LDS stage (32B each)
    const v8h* bs = &Bs[s][lane * 2];
    v16h b0 = *(const v16h*)(bs);
    v16h b1 = *(const v16h*)(bs + 64);
    v16h b2 = *(const v16h*)(bs + 128);
    v16h b3 = *(const v16h*)(bs + 192);

    acc0 = wmma_f16(a, b0, acc0);
    acc1 = wmma_f16(a, b1, acc1);
    acc2 = wmma_f16(a, b2, acc2);
    acc3 = wmma_f16(a, b3, acc3);

    // late ds_store of next stage: s_wait_loadcnt covered by the WMMAs
    Bs[s ^ 1][tid] = stage;
    __syncthreads();
  }
  // peeled final k-step: no reloads, no stores
  {
    v16h a;
    for (int i = 0; i < 8; ++i) { a[i] = a_lo[i]; a[8 + i] = a_hi[i]; }
    const v8h* bs = &Bs[s][lane * 2];
    v16h b0 = *(const v16h*)(bs);
    v16h b1 = *(const v16h*)(bs + 64);
    v16h b2 = *(const v16h*)(bs + 128);
    v16h b3 = *(const v16h*)(bs + 192);
    acc0 = wmma_f16(a, b0, acc0);
    acc1 = wmma_f16(a, b1, acc1);
    acc2 = wmma_f16(a, b2, acc2);
    acc3 = wmma_f16(a, b3, acc3);
  }

  v8f accs[4] = {acc0, acc1, acc2, acc3};
  for (int j = 0; j < 4; ++j) {
    int cc = n0 + j * 16 + l15;
    for (int v = 0; v < 8; ++v) {
      int rr = m0 + hi8 + v;
      size_t idx = (size_t)rr * Nn + cc;
      float val = accs[j][v];
      if (flags & GF_ACC)  val += C[idx];
      if (flags & GF_BIAS) val += bias[cc];
      if (flags & GF_GELU) val = 0.5f * val * (1.f + erff(val * 0.70710678118654752f));
      if (flags & GF_RES)  val += Res[idx];
      C[idx] = val;
    }
  }
}

// --------------------------- flash attention -------------------------------
// head dim 32 == one WMMA K-step. Wave = one (head, 16-query tile).
// Per 32-key block: 2 WMMA for S = Q K^T, online softmax (shfl_xor row
// reductions within 16-lane halves), P transposed through LDS, 2 WMMA for PV.
// K fragments dead-write reloaded right after their S-WMMAs; final key block
// peeled so the hot loop has no guards.
__global__ void attn_kernel(const _Float16* __restrict__ qkvh,
                            const _Float16* __restrict__ vt,
                            float* __restrict__ attnout) {
  __shared__ __attribute__((aligned(32))) _Float16 Pl[4][16 * 32];
  int lane = threadIdx.x & 31;
  int wid = threadIdx.x >> 5;
  int gw = blockIdx.x * 4 + wid;
  int head = gw >> 8;          // N/16 = 256 q-tiles per head
  int q0 = (gw & 255) << 4;
  int hi8 = (lane >> 4) << 3;
  int l15 = lane & 15;

  // Q fragment (A layout): row q0+l15, cols head*32 + {2*hi8..2*hi8+15}
  const _Float16* qp = qkvh + (size_t)(q0 + l15) * 1536 + head * 32;
  v8h qa = *(const v8h*)(qp + 2 * hi8);
  v8h qb = *(const v8h*)(qp + 8 + 2 * hi8);
  v16h aq;
  for (int i = 0; i < 8; ++i) { aq[i] = qa[i]; aq[8 + i] = qb[i]; }

  float m_i[8], l_i[8];
  v8f o0 = {}, o1 = {};
  for (int v = 0; v < 8; ++v) { m_i[v] = -1e30f; l_i[v] = 0.f; }
  const float scale = 0.17677669529663687f;  // 1/sqrt(32)

  // per-lane K fragment base (key row kb + l15): contiguous 32B
  const _Float16* kbase =
      qkvh + (size_t)l15 * 1536 + 512 + head * 32 + ((lane >> 4) << 4);
  const _Float16* vbase =
      vt + (size_t)(head * 32 + l15) * NN + ((lane >> 4) << 4);

  // online softmax + P transpose + PV WMMAs for one 32-key block
  auto process = [&](int kb, v8f s0, v8f s1) {
    float mb[8], p0a[8], p1a[8], lb[8];
    for (int v = 0; v < 8; ++v) {
      s0[v] *= scale; s1[v] *= scale;
      mb[v] = fmaxf(s0[v], s1[v]);
    }
    for (int msk = 1; msk < 16; msk <<= 1)
      for (int v = 0; v < 8; ++v)
        mb[v] = fmaxf(mb[v], __shfl_xor(mb[v], msk, 32));
    for (int v = 0; v < 8; ++v) {
      float mn = fmaxf(m_i[v], mb[v]);
      float corr = __expf(m_i[v] - mn);
      p0a[v] = __expf(s0[v] - mn);
      p1a[v] = __expf(s1[v] - mn);
      lb[v] = p0a[v] + p1a[v];
      m_i[v] = mn;
      l_i[v] *= corr;
      o0[v] *= corr;
      o1[v] *= corr;
    }
    for (int msk = 1; msk < 16; msk <<= 1)
      for (int v = 0; v < 8; ++v) lb[v] += __shfl_xor(lb[v], msk, 32);
    for (int v = 0; v < 8; ++v) l_i[v] += lb[v];

    // transpose P (C layout -> A layout) through LDS
    __syncthreads();
    for (int v = 0; v < 8; ++v) {
      Pl[wid][(v + hi8) * 32 + l15]      = (_Float16)p0a[v];
      Pl[wid][(v + hi8) * 32 + 16 + l15] = (_Float16)p1a[v];
    }
    __syncthreads();
    v8h pa0 = *(const v8h*)&Pl[wid][l15 * 32 + 2 * hi8];
    v8h pa1 = *(const v8h*)&Pl[wid][l15 * 32 + 8 + 2 * hi8];
    v16h ap;
    for (int i = 0; i < 8; ++i) { ap[i] = pa0[i]; ap[8 + i] = pa1[i]; }

    // B = V fragments from pre-transposed vt[(head*32+hd)*N + key]
    const _Float16* vp0 = vbase + kb;
    v16h bv0 = *(const v16h*)vp0;
    v16h bv1 = *(const v16h*)(vp0 + (size_t)16 * NN);
    o0 = wmma_f16(ap, bv0, o0);
    o1 = wmma_f16(ap, bv1, o1);
  };

  // prologue: K fragments for kb = 0
  v16h bk0 = *(const v16h*)(kbase);
  v16h bk1 = *(const v16h*)(kbase + (size_t)16 * 1536);

  int kb = 0;
  for (; kb + 32 < NN; kb += 32) {
    v8f z = {};
    v8f s0 = wmma_f16(aq, bk0, z);
    v8f s1 = wmma_f16(aq, bk1, z);
    // dead-write reload K fragments for the next block (unconditional)
    const _Float16* kp = kbase + (size_t)(kb + 32) * 1536;
    bk0 = *(const v16h*)(kp);
    bk1 = *(const v16h*)(kp + (size_t)16 * 1536);
    __builtin_prefetch(vbase + kb + 32, 0, 3);         // global_prefetch next V
    __builtin_prefetch(vbase + (size_t)16 * NN + kb + 32, 0, 3);
    process(kb, s0, s1);
  }
  // peeled final key block
  {
    v8f z = {};
    v8f s0 = wmma_f16(aq, bk0, z);
    v8f s1 = wmma_f16(aq, bk1, z);
    process(kb, s0, s1);
  }

  for (int v = 0; v < 8; ++v) {
    float inv = 1.f / l_i[v];
    size_t rbase = (size_t)(q0 + hi8 + v) * DD + head * 32;
    attnout[rbase + l15]      = o0[v] * inv;
    attnout[rbase + 16 + l15] = o1[v] * inv;
  }
}

// ------------------------------- host side ---------------------------------

extern "C" void kernel_launch(void* const* d_in, const int* in_sizes, int n_in,
                              void* d_out, int out_size, void* d_ws, size_t ws_size,
                              hipStream_t stream) {
  const float* x        = (const float*)d_in[0];
  const int*   ei       = (const int*)d_in[1];
  const int*   row      = ei;
  const int*   col      = ei + EE;
  const float* cheb_w   = (const float*)d_in[3];
  const float* ln1g     = (const float*)d_in[4];
  const float* ln1b     = (const float*)d_in[5];
  const float* ln2g     = (const float*)d_in[6];
  const float* ln2b     = (const float*)d_in[7];
  const float* ln3g     = (const float*)d_in[8];
  const float* ln3b     = (const float*)d_in[9];
  const float* in_w     = (const float*)d_in[10];
  const float* in_b     = (const float*)d_in[11];
  const float* out_w    = (const float*)d_in[12];
  const float* out_b    = (const float*)d_in[13];
  const float* f1_w     = (const float*)d_in[14];
  const float* f1_b     = (const float*)d_in[15];
  const float* f2_w     = (const float*)d_in[16];
  const float* f2_b     = (const float*)d_in[17];
  float* out = (float*)d_out;

  // ---- carve workspace ----
  char* p = (char*)d_ws;
  auto carve = [&](size_t bytes) -> void* {
    void* r = (void*)p;
    p += (bytes + 255) & ~(size_t)255;
    return r;
  };
  const size_t ND = (size_t)NN * DD;
  float*    h       = (float*)carve(ND * 4);
  float*    txB     = (float*)carve(ND * 4);
  float*    txC     = (float*)carve(ND * 4);
  float*    acc     = (float*)carve(ND * 4);
  float*    x1      = (float*)carve(ND * 4);
  float*    x2      = (float*)carve(ND * 4);
  float*    qkv     = (float*)carve((size_t)NN * 1536 * 4);
  _Float16* qkvh    = (_Float16*)carve((size_t)NN * 1536 * 2);
  _Float16* vt      = (_Float16*)carve((size_t)512 * NN * 2);
  float*    attnout = (float*)carve(ND * 4);
  float*    ffnh    = (float*)carve((size_t)NN * 2048 * 4);
  _Float16* a16     = (_Float16*)carve((size_t)NN * 2048 * 2);  // gemm A (f16)
  _Float16* bp_cheb = (_Float16*)carve((size_t)KC * DD * DD * 2);
  _Float16* bp_in   = (_Float16*)carve((size_t)512 * 1536 * 2);
  _Float16* bp_out  = (_Float16*)carve((size_t)512 * 512 * 2);
  _Float16* bp_f1   = (_Float16*)carve((size_t)512 * 2048 * 2);
  _Float16* bp_f2   = (_Float16*)carve((size_t)2048 * 512 * 2);
  float*    deg     = (float*)carve((size_t)NN * 4);
  float*    dinv    = (float*)carve((size_t)NN * 4);
  float*    nrm     = (float*)carve((size_t)EE * 4);

  const int TB = 256;
  auto blk1 = [](long long n, int t) { return (unsigned)((n + t - 1) / t); };
  auto cvt = [&](const float* src, _Float16* dst, long long n) {
    cvt_f16_kernel<<<blk1(n, TB), TB, 0, stream>>>(src, dst, (int)n);
  };

  // ---- regularizer ----
  hipMemsetAsync(out + ND, 0, sizeof(float), stream);
  reg_loss_kernel<<<blk1(EE, TB), TB, 0, stream>>>(x, row, col, out + ND);

  // ---- edge normalization ----
  hipMemsetAsync(deg, 0, NN * 4, stream);
  deg_kernel<<<blk1(EE, TB), TB, 0, stream>>>(row, deg);
  dinv_kernel<<<blk1(NN, TB), TB, 0, stream>>>(deg, dinv);
  edge_norm_kernel<<<blk1(EE, TB), TB, 0, stream>>>(row, col, dinv, nrm);

  // ---- pack all weights into WMMA B-fragment layout ----
  for (int k = 0; k < KC; ++k)
    pack_b_kernel<<<blk1((size_t)DD * DD / 16, TB), TB, 0, stream>>>(
        cheb_w + (size_t)k * DD * DD, bp_cheb + (size_t)k * DD * DD,
        DD, DD, DD, 0);
  pack_b_kernel<<<blk1((size_t)512 * 1536 / 16, TB), TB, 0, stream>>>(
      in_w, bp_in, 512, 1536, 512, 1);
  pack_b_kernel<<<blk1((size_t)512 * 512 / 16, TB), TB, 0, stream>>>(
      out_w, bp_out, 512, 512, 512, 1);
  pack_b_kernel<<<blk1((size_t)512 * 2048 / 16, TB), TB, 0, stream>>>(
      f1_w, bp_f1, 512, 2048, 512, 1);
  pack_b_kernel<<<blk1((size_t)2048 * 512 / 16, TB), TB, 0, stream>>>(
      f2_w, bp_f2, 2048, 512, 2048, 1);

  // ---- Chebyshev spectral conv ----
  ln_kernel<<<NN, 128, 0, stream>>>(x, ln1g, ln1b, h);

  dim3 gDxD(NN / 128, DD / 64);
  cvt(h, a16, ND);
  gemm_wmma_kernel<<<gDxD, 256, 0, stream>>>(a16, bp_cheb, nullptr, nullptr, acc,
                                             NN, DD, DD, 0);
  hipMemsetAsync(txB, 0, ND * 4, stream);
  prop_kernel<<<blk1((long long)EE * 128, TB), TB, 0, stream>>>(h, txB, row, col,
                                                                nrm, 1.0f);
  cvt(txB, a16, ND);
  gemm_wmma_kernel<<<gDxD, 256, 0, stream>>>(a16, bp_cheb + (size_t)DD * DD,
                                             nullptr, nullptr, acc, NN, DD, DD,
                                             GF_ACC);
  float* p0 = h;  float* p1 = txB;  float* p2 = txC;
  for (int k = 2; k < KC; ++k) {
    neg_copy_kernel<<<blk1(ND, TB), TB, 0, stream>>>(p0, p2, (int)ND);
    prop_kernel<<<blk1((long long)EE * 128, TB), TB, 0, stream>>>(p1, p2, row,
                                                                  col, nrm, 2.0f);
    cvt(p2, a16, ND);
    gemm_wmma_kernel<<<gDxD, 256, 0, stream>>>(a16, bp_cheb + (size_t)k * DD * DD,
                                               nullptr, nullptr, acc, NN, DD, DD,
                                               GF_ACC);
    float* t = p0; p0 = p1; p1 = p2; p2 = t;
  }
  add_kernel<<<blk1(ND, TB), TB, 0, stream>>>(x, acc, x1, (int)ND);

  // ---- attention ----
  ln_kernel<<<NN, 128, 0, stream>>>(x1, ln2g, ln2b, h);
  cvt(h, a16, ND);
  dim3 gQKV(NN / 128, 1536 / 64);
  gemm_wmma_kernel<<<gQKV, 256, 0, stream>>>(a16, bp_in, in_b, nullptr, qkv,
                                             NN, 1536, 512, GF_BIAS);
  cvt(qkv, qkvh, (long long)NN * 1536);
  v_transpose_kernel<<<blk1((size_t)NN * 512, TB), TB, 0, stream>>>(qkv, vt);
  attn_kernel<<<(HH * (NN / 16)) / 4, 128, 0, stream>>>(qkvh, vt, attnout);
  cvt(attnout, a16, ND);
  gemm_wmma_kernel<<<gDxD, 256, 0, stream>>>(a16, bp_out, out_b, x1, x2,
                                             NN, 512, 512, GF_BIAS | GF_RES);

  // ---- FFN ----
  ln_kernel<<<NN, 128, 0, stream>>>(x2, ln3g, ln3b, h);
  cvt(h, a16, ND);
  dim3 gF1(NN / 128, 2048 / 64);
  gemm_wmma_kernel<<<gF1, 256, 0, stream>>>(a16, bp_f1, f1_b, nullptr, ffnh,
                                            NN, 2048, 512, GF_BIAS | GF_GELU);
  cvt(ffnh, a16, (long long)NN * 2048);
  gemm_wmma_kernel<<<gDxD, 256, 0, stream>>>(a16, bp_f2, f2_b, x2, out,
                                             NN, 512, 2048, GF_BIAS | GF_RES);
}
